// LSTM_84490596647096
// MI455X (gfx1250) — matrix operand order repaired and checked by reading, not compiled, not run on hardware
//
#include <hip/hip_runtime.h>

typedef __bf16 bf16;
typedef bf16  v16bf __attribute__((ext_vector_type(16)));
typedef bf16  v8bf  __attribute__((ext_vector_type(8)));
typedef float v8f   __attribute__((ext_vector_type(8)));

#define B_   64
#define T_   256
#define CIN_ 512
#define H_   1024
#define NT_  256   /* 4096/16 N-tiles total */

__device__ __forceinline__ float fsig(float x) {
    return 1.0f / (1.0f + __expf(-x));
}
__device__ __forceinline__ float ftanh_(float x) {
    float e = __expf(-2.0f * fabsf(x));        // e in (0,1], no overflow
    float t = (1.0f - e) / (1.0f + e);
    return copysignf(t, x);
}

// ---------------- one-time prep kernels ----------------

__global__ void k_bias_sum(const float* __restrict__ a, const float* __restrict__ b,
                           float* __restrict__ o, int n) {
    int i = blockIdx.x * blockDim.x + threadIdx.x;
    if (i < n) o[i] = a[i] + b[i];
}

__global__ void k_f32_to_bf16(const float* __restrict__ s, bf16* __restrict__ d, int n) {
    int i = blockIdx.x * blockDim.x + threadIdx.x;
    if (i < n) d[i] = (bf16)s[i];
}

// Pack W [4096 x Kw] (row-major over k) into WMMA-B order:
// dst[ntile][kstep][lane][e] , element e of lane = W[n][k] with
//   n = ntile*16 + (lane&15),  k = kstep*32 + (lane>>4)*16 + e
// (contiguous k order chosen here, same as A: lanes 0-15 hold K=0..15, lanes 16-31 K=16..31)
__global__ void k_pack_w(const float* __restrict__ W, int Kw, bf16* __restrict__ dst,
                         int ksteps_total, int kstep0) {
    int idx  = blockIdx.x * blockDim.x + threadIdx.x;
    int lane = idx & 31;
    int t    = idx >> 5;
    int kst  = Kw >> 5;                 // local ksteps in this matrix
    int ksl  = t % kst;
    int nt   = t / kst;
    if (nt >= NT_) return;
    int n  = nt * 16 + (lane & 15);
    int ks = ksl * 32 + ((lane >> 4) << 4);
    const float* src = W + (size_t)n * Kw + ks;   // 16 contiguous floats
    bf16* d = dst + ((((size_t)nt * ksteps_total + kstep0 + ksl) * 32 + lane) << 4);
#pragma unroll
    for (int e = 0; e < 16; ++e) d[e] = (bf16)src[e];
}

// ---------------- per-step fused GEMM + LSTM cell ----------------

// One phase of the K loop: A streamed from a single source (simple pointer
// induction, no per-load select), 4 gate WMMAs share each A operand.
template<int NKS>
__device__ __forceinline__ void gemm_phase(
    const bf16* __restrict__ pa,
    const v16bf*& pb0, const v16bf*& pb1, const v16bf*& pb2, const v16bf*& pb3,
    v8f& acc0, v8f& acc1, v8f& acc2, v8f& acc3)
{
#pragma unroll 4
    for (int ks = 0; ks < NKS; ++ks) {
        // A operand in this code's contiguous k order (same as packed B): lane holds K=[k0,k0+16)
        v8bf lo = *(const v8bf*)(pa);
        v8bf hi = *(const v8bf*)(pa + 8);
        v16bf a;
#pragma unroll
        for (int e = 0; e < 8; ++e) { a[e] = lo[e]; a[8 + e] = hi[e]; }

        v16bf b0 = *pb0; pb0 += 32;
        v16bf b1 = *pb1; pb1 += 32;
        v16bf b2 = *pb2; pb2 += 32;
        v16bf b3 = *pb3; pb3 += 32;

        acc0 = __builtin_amdgcn_wmma_f32_16x16x32_bf16(false, a, false, b0, (short)0, acc0, false, false);
        acc1 = __builtin_amdgcn_wmma_f32_16x16x32_bf16(false, a, false, b1, (short)0, acc1, false, false);
        acc2 = __builtin_amdgcn_wmma_f32_16x16x32_bf16(false, a, false, b2, (short)0, acc2, false, false);
        acc3 = __builtin_amdgcn_wmma_f32_16x16x32_bf16(false, a, false, b3, (short)0, acc3, false, false);

        pa += 32;
    }
}

// gates = [Asrc0 | Asrc1] @ Wp + bias ; then i,f,g,o -> c,h update.
// grid = 32 blocks x 256 threads = 256 waves = 4 mtiles x 64 jtiles.
// KS0/KS1 = k-steps (of 32) drawn from a0 / a1; LDA0 = row stride of a0.
template<int KS0, int KS1, int LDA0>
__global__ __launch_bounds__(256) void k_lstm_step(
    const bf16* __restrict__ a0,
    const bf16* __restrict__ a1,
    const bf16* __restrict__ Wp,
    const float* __restrict__ bias,
    float* __restrict__ cst,
    float* __restrict__ hf,
    bf16*  __restrict__ hbf_out)
{
    constexpr int KSTEPS = KS0 + KS1;

    const int lane  = threadIdx.x & 31;
    const int w     = (blockIdx.x << 3) + (threadIdx.x >> 5);   // 0..255
    const int mtile = w >> 6;                                   // 0..3
    const int jtile = w & 63;                                   // 0..63
    const int arow  = (mtile << 4) + (lane & 15);
    const int khalf = (lane >> 4) << 4;                         // 0 or 16

    v8f acc0 = {}, acc1 = {}, acc2 = {}, acc3 = {};
    const v16bf* bp  = (const v16bf*)Wp;
    const v16bf* pb0 = bp + ((size_t)(jtile +   0) * KSTEPS) * 32 + lane;
    const v16bf* pb1 = bp + ((size_t)(jtile +  64) * KSTEPS) * 32 + lane;
    const v16bf* pb2 = bp + ((size_t)(jtile + 128) * KSTEPS) * 32 + lane;
    const v16bf* pb3 = bp + ((size_t)(jtile + 192) * KSTEPS) * 32 + lane;

    gemm_phase<KS0>(a0 + (size_t)arow * LDA0 + khalf, pb0, pb1, pb2, pb3,
                    acc0, acc1, acc2, acc3);
    gemm_phase<KS1>(a1 + (size_t)arow * H_ + khalf, pb0, pb1, pb2, pb3,
                    acc0, acc1, acc2, acc3);

    // epilogue: this wave owns all 4 gates of columns jtile*16+(lane&15)
    const int   col = (jtile << 4) + (lane & 15);
    const float bi = bias[col];
    const float bf = bias[H_     + col];
    const float bg = bias[2 * H_ + col];
    const float bo = bias[3 * H_ + col];
    const int rbase = (mtile << 4) + ((lane >> 4) << 3);  // C/D: VGPR v -> row v / v+8

#pragma unroll
    for (int v = 0; v < 8; ++v) {
        const int    m   = rbase + v;
        const size_t off = (size_t)m * H_ + col;
        float iv = fsig(acc0[v] + bi);
        float fv = fsig(acc1[v] + bf);
        float gv = ftanh_(acc2[v] + bg);
        float ov = fsig(acc3[v] + bo);
        float cn = fv * cst[off] + iv * gv;
        float hn = ov * ftanh_(cn);
        cst[off]     = cn;
        hf[off]      = hn;
        hbf_out[off] = (bf16)hn;
    }
}

// ---------------- host side ----------------

extern "C" void kernel_launch(void* const* d_in, const int* in_sizes, int n_in,
                              void* d_out, int out_size, void* d_ws, size_t ws_size,
                              hipStream_t stream)
{
    (void)in_sizes; (void)n_in; (void)out_size; (void)ws_size;

    const float* x    = (const float*)d_in[0];
    const float* Wih0 = (const float*)d_in[1];
    const float* Whh0 = (const float*)d_in[2];
    const float* bih0 = (const float*)d_in[3];
    const float* bhh0 = (const float*)d_in[4];
    const float* Wih1 = (const float*)d_in[5];
    const float* Whh1 = (const float*)d_in[6];
    const float* bih1 = (const float*)d_in[7];
    const float* bhh1 = (const float*)d_in[8];

    const int KST0 = (CIN_ + H_) / 32;   // 48
    const int KST1 = (2 * H_) / 32;      // 64

    char* ws = (char*)d_ws;
    bf16* w0p = (bf16*)ws;  ws += sizeof(bf16)  * (size_t)NT_ * KST0 * 32 * 16;   // 12.6 MB
    bf16* w1p = (bf16*)ws;  ws += sizeof(bf16)  * (size_t)NT_ * KST1 * 32 * 16;   // 16.8 MB
    bf16* xbf = (bf16*)ws;  ws += sizeof(bf16)  * (size_t)T_ * B_ * CIN_;         // 16.8 MB
    float* b0 = (float*)ws; ws += sizeof(float) * 4 * H_;
    float* b1 = (float*)ws; ws += sizeof(float) * 4 * H_;
    char* states = ws;
    float* c0f = (float*)ws; ws += sizeof(float) * B_ * H_;
    float* c1f = (float*)ws; ws += sizeof(float) * B_ * H_;
    float* h0f = (float*)ws; ws += sizeof(float) * B_ * H_;
    float* h1f = (float*)ws; ws += sizeof(float) * B_ * H_;
    bf16* h0b[2]; bf16* h1b[2];
    h0b[0] = (bf16*)ws; ws += sizeof(bf16) * B_ * H_;
    h0b[1] = (bf16*)ws; ws += sizeof(bf16) * B_ * H_;
    h1b[0] = (bf16*)ws; ws += sizeof(bf16) * B_ * H_;
    h1b[1] = (bf16*)ws; ws += sizeof(bf16) * B_ * H_;
    const size_t states_bytes = (size_t)(ws - states);

    // 1) pack weights to WMMA-B bf16 layout (L2-resident thereafter)
    {
        int thr, blk;
        thr = NT_ * (CIN_ / 32) * 32; blk = (thr + 255) / 256;
        k_pack_w<<<blk, 256, 0, stream>>>(Wih0, CIN_, w0p, KST0, 0);
        thr = NT_ * (H_ / 32) * 32;   blk = (thr + 255) / 256;
        k_pack_w<<<blk, 256, 0, stream>>>(Whh0, H_, w0p, KST0, CIN_ / 32);
        k_pack_w<<<blk, 256, 0, stream>>>(Wih1, H_, w1p, KST1, 0);
        k_pack_w<<<blk, 256, 0, stream>>>(Whh1, H_, w1p, KST1, H_ / 32);
    }
    // 2) combined biases
    k_bias_sum<<<(4 * H_ + 255) / 256, 256, 0, stream>>>(bih0, bhh0, b0, 4 * H_);
    k_bias_sum<<<(4 * H_ + 255) / 256, 256, 0, stream>>>(bih1, bhh1, b1, 4 * H_);
    // 3) x -> bf16 once
    {
        int n = T_ * B_ * CIN_;
        k_f32_to_bf16<<<(n + 255) / 256, 256, 0, stream>>>(x, xbf, n);
    }
    // 4) zero states (h, c, and both h-bf16 buffers)
    hipMemsetAsync(states, 0, states_bytes, stream);

    // 5) recurrence: 2 fused GEMM+cell kernels per timestep
    for (int t = 0; t < T_; ++t) {
        const int rb = t & 1, wb = rb ^ 1;
        // layer 0: A = [x_t | h0] , K = 512 + 1024
        k_lstm_step<16, 32, CIN_><<<32, 256, 0, stream>>>(
            xbf + (size_t)t * B_ * CIN_, h0b[rb],
            w0p, b0, c0f, h0f, h0b[wb]);
        // layer 1: A = [h0(new) | h1] , K = 1024 + 1024
        k_lstm_step<32, 32, H_><<<32, 256, 0, stream>>>(
            h0b[wb], h1b[rb],
            w1p, b1, c1f, h1f, h1b[wb]);
    }

    // 6) final h1 (fp32) -> d_out
    hipMemcpyAsync(d_out, h1f, sizeof(float) * B_ * H_, hipMemcpyDeviceToDevice, stream);
}
